// Hawk_35442070127170
// MI455X (gfx1250) — matrix operand order repaired and checked by reading, not compiled, not run on hardware
//
#include <hip/hip_runtime.h>
#include <hip/hip_bf16.h>
#include <math.h>

typedef __attribute__((ext_vector_type(16))) _Float16 v16h;
typedef __attribute__((ext_vector_type(8)))  float    v8f;
typedef __attribute__((ext_vector_type(4)))  float    v4f;

// Model dims (match the reference)
constexpr int BATCH = 2;
constexpr int SEQ   = 1024;
constexpr int DIM   = 1024;
constexpr int RNN   = 1536;
constexpr int EXP   = 3 * DIM;   // 3072
constexpr int VOC   = 32000;
constexpr int NLAY  = 4;
constexpr int MTOK  = BATCH * SEQ;  // 2048 rows

// ---------------- async global -> LDS helpers (gfx1250) ----------------
__device__ __forceinline__ void async_b128(uint32_t lds_addr, const float* gsrc)
{
    // GLOBAL_LOAD_ASYNC_TO_LDS_B128: vdst = LDS byte address, vaddr = 64-bit global
    asm volatile("global_load_async_to_lds_b128 %0, %1, off"
                 :: "v"(lds_addr), "v"((unsigned long long)(uintptr_t)gsrc)
                 : "memory");
}

template<int N>
__device__ __forceinline__ void wait_async()
{
#if __has_builtin(__builtin_amdgcn_s_wait_asynccnt)
    __builtin_amdgcn_s_wait_asynccnt((short)N);
#else
    asm volatile("s_wait_asynccnt %0" :: "i"(N) : "memory");
#endif
}

// ---------------- GEMM: C[M,N] = act(A[M,K] @ W[N,K]^T + bias) (+ Cin) --------
// f32 tiles DMA'd into LDS via async loads (double-buffered), converted to f16
// at fragment build, accumulated with v_wmma_f32_16x16x32_f16.
constexpr int BM = 64, BN = 128, BK = 32;
constexpr int ASTF = BK + 4;   // padded LDS row stride (floats): 144B, 16B aligned
constexpr int BSTF = BK + 4;

template<int ACT>
__device__ __forceinline__ float act_apply(float x) {
    if constexpr (ACT == 1) {                 // SiLU
        return x / (1.0f + __expf(-x));
    } else if constexpr (ACT == 2) {          // GELU (tanh approx, jax default)
        float x3 = x * x * x;
        return 0.5f * x * (1.0f + tanhf(0.7978845608028654f * (x + 0.044715f * x3)));
    } else if constexpr (ACT == 3) {          // sigmoid
        return 1.0f / (1.0f + __expf(-x));
    } else {
        return x;
    }
}

template<int ACT, bool ADDC>
__global__ __launch_bounds__(256)
void gemm_wmma_kernel(const float* __restrict__ A, const float* __restrict__ W,
                      const float* __restrict__ bias, const float* __restrict__ Cin,
                      float* __restrict__ C, int M, int N, int K)
{
    __shared__ __align__(16) float As[2][BM * ASTF];
    __shared__ __align__(16) float Bs[2][BN * BSTF];

    const int tid = threadIdx.x;
    const int bm  = blockIdx.y * BM;
    const int bn  = blockIdx.x * BN;
    const int w   = tid >> 5;              // wave 0..7
    const int l   = tid & 31;              // lane
    const int wm  = (w >> 2) * 32;         // wave M offset within block tile
    const int wn  = (w & 3) * 32;          // wave N offset
    const int lm  = l & 15;
    const int lh  = l >> 4;

    v8f acc[2][2] = {};

    // Per-thread staging: A tile 64x32 f32 = 512 16B chunks (2/thread),
    //                     B tile 128x32 f32 = 1024 chunks (4/thread).
    auto issue_tile = [&](int k0, int buf) {
        #pragma unroll
        for (int i = 0; i < 2; ++i) {
            int idx = tid + i * 256;
            int row = idx >> 3;
            int kc  = (idx & 7) * 4;
            async_b128((uint32_t)(uintptr_t)&As[buf][row * ASTF + kc],
                       A + (size_t)(bm + row) * K + k0 + kc);
        }
        #pragma unroll
        for (int i = 0; i < 4; ++i) {
            int idx = tid + i * 256;
            int row = idx >> 3;
            int kc  = (idx & 7) * 4;
            async_b128((uint32_t)(uintptr_t)&Bs[buf][row * BSTF + kc],
                       W + (size_t)(bn + row) * K + k0 + kc);
        }
    };

    const int nk = K / BK;
    issue_tile(0, 0);

    for (int t = 0; t < nk; ++t) {
        const int buf = t & 1;
        if (t + 1 < nk) {
            issue_tile((t + 1) * BK, buf ^ 1);
            wait_async<6>();      // retire tile t (in-order), tile t+1 in flight
        } else {
            wait_async<0>();
        }
        __syncthreads();

        // ---- build f16 fragments (packed cvt) per ISA wave32 layouts ----
        union FragH { v16h v; _Float16 h[16]; };
        FragH af[2], bf[2];
        #pragma unroll
        for (int i = 0; i < 2; ++i) {
            // A 16x32: lane holds K chunks [8h..8h+7] and [16+8h..16+8h+7] of row lm
            const float* p = &As[buf][(wm + i * 16 + lm) * ASTF + 8 * lh];
            v4f x0 = *(const v4f*)(p);
            v4f x1 = *(const v4f*)(p + 4);
            v4f x2 = *(const v4f*)(p + 16);
            v4f x3 = *(const v4f*)(p + 20);
            #pragma unroll
            for (int e = 0; e < 4; ++e) {
                af[i].h[e]      = (_Float16)x0[e];
                af[i].h[4 + e]  = (_Float16)x1[e];
                af[i].h[8 + e]  = (_Float16)x2[e];
                af[i].h[12 + e] = (_Float16)x3[e];
            }
        }
        #pragma unroll
        for (int j = 0; j < 2; ++j) {
            // B 32x16: lane holds K chunks [8h..8h+7] and [16+8h..16+8h+7] of column lm
            const float* p = &Bs[buf][(wn + j * 16 + lm) * BSTF + 8 * lh];
            v4f x0 = *(const v4f*)(p);
            v4f x1 = *(const v4f*)(p + 4);
            v4f x2 = *(const v4f*)(p + 16);
            v4f x3 = *(const v4f*)(p + 20);
            #pragma unroll
            for (int e = 0; e < 4; ++e) {
                bf[j].h[e]      = (_Float16)x0[e];
                bf[j].h[4 + e]  = (_Float16)x1[e];
                bf[j].h[8 + e]  = (_Float16)x2[e];
                bf[j].h[12 + e] = (_Float16)x3[e];
            }
        }

        #pragma unroll
        for (int i = 0; i < 2; ++i)
            #pragma unroll
            for (int j = 0; j < 2; ++j)
                acc[i][j] = __builtin_amdgcn_wmma_f32_16x16x32_f16(
                    false, af[i].v, false, bf[j].v, (short)0, acc[i][j], false, false);
        __syncthreads();
    }

    // ---- epilogue: bias + activation (+ residual) ----
    #pragma unroll
    for (int i = 0; i < 2; ++i) {
        #pragma unroll
        for (int j = 0; j < 2; ++j) {
            int col = bn + wn + j * 16 + lm;
            float bv = bias[col];
            #pragma unroll
            for (int r = 0; r < 8; ++r) {
                int row = bm + wm + i * 16 + lh * 8 + r;
                float v = acc[i][j][r] + bv;
                v = act_apply<ACT>(v);
                if (ADDC) v += Cin[(size_t)row * N + col];
                C[(size_t)row * N + col] = v;
            }
        }
    }
}

// ---------------- elementwise / glue kernels ----------------

__global__ __launch_bounds__(256)
void embed_kernel(const int* __restrict__ x, const float* __restrict__ tok,
                  const float* __restrict__ pos, float* __restrict__ out)
{
    int bs = blockIdx.x;                 // 0..MTOK-1
    int s  = bs % SEQ;
    int t  = x[bs];
    for (int d = threadIdx.x; d < DIM; d += 256)
        out[(size_t)bs * DIM + d] = tok[(size_t)t * DIM + d] + pos[(size_t)s * DIM + d];
}

__global__ __launch_bounds__(256)
void rmsnorm_kernel(const float* __restrict__ x, const float* __restrict__ w,
                    float* __restrict__ out, int ncols)
{
    __shared__ float red[8];
    __shared__ float stot;
    int row = blockIdx.x;
    const float* xr = x + (size_t)row * ncols;
    float ss = 0.f;
    for (int d = threadIdx.x; d < ncols; d += 256) { float v = xr[d]; ss += v * v; }
    #pragma unroll
    for (int off = 16; off; off >>= 1) ss += __shfl_xor(ss, off, 32);   // wave32
    if ((threadIdx.x & 31) == 0) red[threadIdx.x >> 5] = ss;
    __syncthreads();
    if (threadIdx.x == 0) {
        float t = 0.f;
        #pragma unroll
        for (int i = 0; i < 8; ++i) t += red[i];
        stot = rsqrtf(t / (float)ncols + 1e-6f);
    }
    __syncthreads();
    float sc = stot;
    for (int d = threadIdx.x; d < ncols; d += 256)
        out[(size_t)row * ncols + d] = xr[d] * sc * w[d];
}

// im2col for Conv1d(k=4, dilation=2, pad=3): K index = r_in*4 + k matches
// conv_w [R_out, R_in, 4] memory layout.  tap time = s - 3 + 2k.
__global__ __launch_bounds__(256)
void im2col_kernel(const float* __restrict__ act, float* __restrict__ im)
{
    size_t idx = (size_t)blockIdx.x * 256 + threadIdx.x;
    size_t total = (size_t)MTOK * 4 * RNN;
    if (idx >= total) return;
    int kk  = (int)(idx % (4 * RNN));
    int bs  = (int)(idx / (4 * RNN));
    int rin = kk >> 2;
    int k   = kk & 3;
    int b   = bs / SEQ, s = bs % SEQ;
    int t   = s - 3 + 2 * k;
    float v = (t >= 0 && t < SEQ) ? act[(size_t)(b * SEQ + t) * RNN + rin] : 0.f;
    im[idx] = v;
}

// r,i already sigmoided by GEMM epilogue. Compute a = exp(-8*softplus(-lam)*r),
// g = sqrt(max(1-a^2,1e-6)) * i * xconv.  Overwrites rbuf->a, ibuf->g.
__global__ __launch_bounds__(256)
void rglru_prep_kernel(float* __restrict__ rbuf, float* __restrict__ ibuf,
                       const float* __restrict__ xconv, const float* __restrict__ lam)
{
    size_t idx = (size_t)blockIdx.x * 256 + threadIdx.x;
    size_t total = (size_t)MTOK * RNN;
    if (idx >= total) return;
    int rch = (int)(idx % RNN);
    float r  = rbuf[idx];
    float la = -8.0f * log1pf(__expf(-lam[rch])) * r;   // log a
    float a  = __expf(la);
    float g  = sqrtf(fmaxf(1.0f - __expf(2.0f * la), 1e-6f)) * ibuf[idx] * xconv[idx];
    rbuf[idx] = a;
    ibuf[idx] = g;
}

// Sequential linear recurrence h_t = a_t*h_{t-1} + g_t, one thread per (b, channel).
__global__ __launch_bounds__(256)
void scan_kernel(const float* __restrict__ a, const float* __restrict__ g,
                 float* __restrict__ h)
{
    int c = blockIdx.x * 256 + threadIdx.x;
    if (c >= BATCH * RNN) return;
    int b = c / RNN, rch = c % RNN;
    size_t base = (size_t)b * SEQ * RNN + rch;
    float hv = 0.f;
    for (int s = 0; s < SEQ; ++s) {
        size_t idx = base + (size_t)s * RNN;
        hv = a[idx] * hv + g[idx];
        h[idx] = hv;
    }
}

__global__ __launch_bounds__(256)
void mul_kernel(float* __restrict__ g, const float* __restrict__ u, size_t n)
{
    size_t idx = (size_t)blockIdx.x * 256 + threadIdx.x;
    if (idx < n) g[idx] *= u[idx];
}

// ---------------- host orchestration ----------------

template<int ACT, bool ADDC>
static void gemm(hipStream_t st, const float* A, const float* W, const float* bias,
                 const float* Cin, float* C, int M, int N, int K)
{
    dim3 grid(N / BN, M / BM), block(256);
    gemm_wmma_kernel<ACT, ADDC><<<grid, block, 0, st>>>(A, W, bias, Cin, C, M, N, K);
}

static inline float* ws_take(void* base, size_t& off, size_t count)
{
    float* p = (float*)((char*)base + off);
    off += ((count * sizeof(float) + 255) & ~(size_t)255);
    return p;
}

extern "C" void kernel_launch(void* const* d_in, const int* in_sizes, int n_in,
                              void* d_out, int out_size, void* d_ws, size_t ws_size,
                              hipStream_t stream)
{
    (void)in_sizes; (void)n_in; (void)out_size; (void)ws_size;
    const int*   x        = (const int*)  d_in[0];
    const float* tok_emb  = (const float*)d_in[1];
    const float* pos_emb  = (const float*)d_in[2];
    const float* norm1_w  = (const float*)d_in[3];
    const float* norm2_w  = (const float*)d_in[4];
    const float* lin1_w   = (const float*)d_in[5];
    const float* lin1_b   = (const float*)d_in[6];
    const float* conv_w   = (const float*)d_in[7];
    const float* conv_b   = (const float*)d_in[8];
    const float* wa       = (const float*)d_in[9];
    const float* ba       = (const float*)d_in[10];
    const float* wx       = (const float*)d_in[11];
    const float* bx       = (const float*)d_in[12];
    const float* lam      = (const float*)d_in[13];
    const float* lin2_w   = (const float*)d_in[14];
    const float* lin2_b   = (const float*)d_in[15];
    const float* mlp_wg   = (const float*)d_in[16];
    const float* mlp_bg   = (const float*)d_in[17];
    const float* mlp_wu   = (const float*)d_in[18];
    const float* mlp_bu   = (const float*)d_in[19];
    const float* mlp_wd   = (const float*)d_in[20];
    const float* mlp_bd   = (const float*)d_in[21];
    const float* out_nw   = (const float*)d_in[22];
    const float* out_w    = (const float*)d_in[23];
    const float* out_b    = (const float*)d_in[24];
    const float* head_w   = (const float*)d_in[25];
    const float* head_b   = (const float*)d_in[26];
    float* logits = (float*)d_out;

    size_t off = 0;
    float* X    = ws_take(d_ws, off, (size_t)MTOK * DIM);
    float* NRM  = ws_take(d_ws, off, (size_t)MTOK * DIM);
    float* ACT1 = ws_take(d_ws, off, (size_t)MTOK * RNN);
    float* IM2  = ws_take(d_ws, off, (size_t)MTOK * 4 * RNN);
    float* CONV = ws_take(d_ws, off, (size_t)MTOK * RNN);
    float* RG   = ws_take(d_ws, off, (size_t)MTOK * RNN);
    float* IG   = ws_take(d_ws, off, (size_t)MTOK * RNN);
    float* HS   = ws_take(d_ws, off, (size_t)MTOK * RNN);
    float* G    = ws_take(d_ws, off, (size_t)MTOK * EXP);
    float* U    = ws_take(d_ws, off, (size_t)MTOK * EXP);
    float* HIN  = ws_take(d_ws, off, (size_t)MTOK * DIM);

    // h = tok_emb[x] + pos_emb
    embed_kernel<<<MTOK, 256, 0, stream>>>(x, tok_emb, pos_emb, X);

    const size_t im2n  = (size_t)MTOK * 4 * RNN;
    const size_t rgn   = (size_t)MTOK * RNN;
    const size_t expn  = (size_t)MTOK * EXP;

    for (int l = 0; l < NLAY; ++l) {
        // --- recurrent branch ---
        rmsnorm_kernel<<<MTOK, 256, 0, stream>>>(X, norm1_w + (size_t)l * DIM, NRM, DIM);
        gemm<1, false>(stream, NRM, lin1_w + (size_t)l * RNN * DIM,
                       lin1_b + (size_t)l * RNN, nullptr, ACT1, MTOK, RNN, DIM);  // SiLU
        im2col_kernel<<<(int)((im2n + 255) / 256), 256, 0, stream>>>(ACT1, IM2);
        gemm<0, false>(stream, IM2, conv_w + (size_t)l * RNN * RNN * 4,
                       conv_b + (size_t)l * RNN, nullptr, CONV, MTOK, RNN, 4 * RNN);
        gemm<3, false>(stream, CONV, wa + (size_t)l * RNN * RNN,
                       ba + (size_t)l * RNN, nullptr, RG, MTOK, RNN, RNN);        // sigmoid
        gemm<3, false>(stream, CONV, wx + (size_t)l * RNN * RNN,
                       bx + (size_t)l * RNN, nullptr, IG, MTOK, RNN, RNN);        // sigmoid
        rglru_prep_kernel<<<(int)((rgn + 255) / 256), 256, 0, stream>>>(
            RG, IG, CONV, lam + (size_t)l * RNN);
        scan_kernel<<<(BATCH * RNN + 255) / 256, 256, 0, stream>>>(RG, IG, HS);
        gemm<0, true>(stream, HS, lin2_w + (size_t)l * DIM * RNN,
                      lin2_b + (size_t)l * DIM, X, X, MTOK, DIM, RNN);            // +residual
        // --- MLP branch ---
        rmsnorm_kernel<<<MTOK, 256, 0, stream>>>(X, norm2_w + (size_t)l * DIM, NRM, DIM);
        gemm<2, false>(stream, NRM, mlp_wg + (size_t)l * EXP * DIM,
                       mlp_bg + (size_t)l * EXP, nullptr, G, MTOK, EXP, DIM);     // GELU
        gemm<0, false>(stream, NRM, mlp_wu + (size_t)l * EXP * DIM,
                       mlp_bu + (size_t)l * EXP, nullptr, U, MTOK, EXP, DIM);
        mul_kernel<<<(int)((expn + 255) / 256), 256, 0, stream>>>(G, U, expn);
        gemm<0, true>(stream, G, mlp_wd + (size_t)l * DIM * EXP,
                      mlp_bd + (size_t)l * DIM, X, X, MTOK, DIM, EXP);            // +residual
    }

    // final norm -> out linear (D->D) -> lm head (D->V)
    rmsnorm_kernel<<<MTOK, 256, 0, stream>>>(X, out_nw, NRM, DIM);
    gemm<0, false>(stream, NRM, out_w, out_b, nullptr, HIN, MTOK, DIM, DIM);
    gemm<0, false>(stream, HIN, head_w, head_b, nullptr, logits, MTOK, VOC, DIM);
}
